// RetinaHead_85255100826255
// MI455X (gfx1250) — matrix operand order repaired and checked by reading, not compiled, not run on hardware
//
#include <hip/hip_runtime.h>
#include <stdint.h>

#define NUM_CLASSES 80
#define NUM_ANCHORS 9
#define CCH 256

typedef __attribute__((ext_vector_type(16))) __bf16 v16bf;
typedef __attribute__((ext_vector_type(8)))  float  v8f;

union Frag16 { v16bf v; uint4 q[2]; };

__device__ __forceinline__ uint16_t f2bf(float f) {
  uint32_t u = __float_as_uint(f);
  u += 0x7FFFu + ((u >> 16) & 1u);   // round-to-nearest-even
  return (uint16_t)(u >> 16);
}

__device__ __forceinline__ v8f wmma_bf16(const Frag16& a, const Frag16& b, v8f c) {
  return __builtin_amdgcn_wmma_f32_16x16x32_bf16(false, a.v, false, b.v,
                                                 (short)0, c, false, false);
}

// ---------------- utility / conversion kernels ----------------

__global__ void k_zero2(float* out) { if (threadIdx.x < 2) out[threadIdx.x] = 0.f; }

__global__ void k_f32_to_bf16(const float* __restrict__ src, uint16_t* __restrict__ dst, int n) {
  for (int i = blockIdx.x * blockDim.x + threadIdx.x; i < n; i += gridDim.x * blockDim.x)
    dst[i] = f2bf(src[i]);
}

// in_proj: src [5][Cin][Cout] -> dst [5][Cout][Cin]
__global__ void k_cvt_inproj(const float* __restrict__ src, uint16_t* __restrict__ dst) {
  int i = blockIdx.x * blockDim.x + threadIdx.x;
  if (i >= 5 * CCH * CCH) return;
  int l = i / (CCH * CCH), r = i % (CCH * CCH), co = r / CCH, ci = r % CCH;
  dst[i] = f2bf(src[(size_t)(l * CCH + ci) * CCH + co]);
}

// conv stack: src [4][3][3][Cin][Cout] HWIO -> dst [4][Cout][K=tap*Cin+ci]
__global__ void k_cvt_convw(const float* __restrict__ src, uint16_t* __restrict__ dst) {
  int i = blockIdx.x * blockDim.x + threadIdx.x;
  if (i >= 4 * CCH * 9 * CCH) return;
  int j  = i / (CCH * 9 * CCH);
  int r  = i % (CCH * 9 * CCH);
  int co = r / (9 * CCH);
  int k  = r % (9 * CCH);
  int tap = k / CCH, ci = k % CCH;
  dst[i] = f2bf(src[(size_t)((j * 9 + tap) * CCH + ci) * CCH + co]);
}

// predictor: src [3][3][Cin][CoutSrc] -> dst [CoutPad][9*Cin], zero padded
__global__ void k_cvt_predw(const float* __restrict__ src, uint16_t* __restrict__ dst,
                            int CoutSrc, int CoutPad) {
  int i = blockIdx.x * blockDim.x + threadIdx.x;
  if (i >= CoutPad * 9 * CCH) return;
  int co = i / (9 * CCH);
  int k  = i % (9 * CCH);
  int tap = k / CCH, ci = k % CCH;
  float v = (co < CoutSrc) ? src[(size_t)(tap * CCH + ci) * CoutSrc + co] : 0.f;
  dst[i] = f2bf(v);
}

// ------- implicit-GEMM conv (Cin=Cout=256); block = 16px tile x 256 out-ch -------
// 4 waves/block, wave w computes channels [64w, 64w+64); A tile staged per-tap in LDS.

template <bool RELU, int TAPS>
__global__ __launch_bounds__(128)
void k_conv_wmma(const uint16_t* __restrict__ Ab,   // [B*H*W][256] bf16
                 const uint16_t* __restrict__ Wt,   // [256][TAPS*256] bf16
                 const float*    __restrict__ bias, // [256]
                 uint16_t*       __restrict__ Ob,   // [B*H*W][256] bf16
                 int H, int W) {
  __shared__ uint16_t smem[16 * CCH];               // 8 KB: 16 px x 256 ch
  const int HW   = H * W;
  const int KTOT = TAPS * CCH;
  int pt   = blockIdx.x;
  int tid  = threadIdx.x;
  int wave = tid >> 5, lane = tid & 31;
  int m = lane & 15, half = lane >> 4;
  int cobase = wave * 64;

  // hoisted staging coordinates: thread handles chunks tid + k*128 (16B each)
  int sb[4], sy[4], sx[4];
#pragma unroll
  for (int k = 0; k < 4; ++k) {
    int c = tid + k * 128;
    int p = (pt << 4) + (c >> 5);
    sb[k] = p / HW; int rem = p % HW; sy[k] = rem / W; sx[k] = rem % W;
  }

  v8f acc0 = {}, acc1 = {}, acc2 = {}, acc3 = {};
  const uint16_t* wbase = Wt + (size_t)(cobase + m) * KTOT + half * 16;

  for (int t = 0; t < TAPS; ++t) {
    int dy = (TAPS == 9) ? (t / 3) - 1 : 0;
    int dx = (TAPS == 9) ? (t % 3) - 1 : 0;
    __syncthreads();                                 // previous-tap consumers done
#pragma unroll
    for (int k = 0; k < 4; ++k) {
      int c   = tid + k * 128;
      int off = (c & 31) * 8;
      int px  = c >> 5;
      int yy = sy[k] + dy, xx = sx[k] + dx;
      uint4 val = make_uint4(0u, 0u, 0u, 0u);
      if (yy >= 0 && yy < H && xx >= 0 && xx < W)
        val = *(const uint4*)(Ab + ((size_t)sb[k] * HW + (size_t)yy * W + xx) * CCH + off);
      *(uint4*)(smem + px * CCH + off) = val;
    }
    __syncthreads();
    const uint16_t* wt = wbase + t * CCH;
#pragma unroll
    for (int kc = 0; kc < CCH; kc += 32) {
      Frag16 a;
      a.q[0] = *(const uint4*)(smem + m * CCH + kc + half * 16);
      a.q[1] = *(const uint4*)(smem + m * CCH + kc + half * 16 + 8);
      const uint16_t* wp = wt + kc;
      Frag16 b0, b1, b2, b3;
      b0.q[0] = *(const uint4*)(wp);                     b0.q[1] = *(const uint4*)(wp + 8);
      b1.q[0] = *(const uint4*)(wp + (size_t)16 * KTOT); b1.q[1] = *(const uint4*)(wp + (size_t)16 * KTOT + 8);
      b2.q[0] = *(const uint4*)(wp + (size_t)32 * KTOT); b2.q[1] = *(const uint4*)(wp + (size_t)32 * KTOT + 8);
      b3.q[0] = *(const uint4*)(wp + (size_t)48 * KTOT); b3.q[1] = *(const uint4*)(wp + (size_t)48 * KTOT + 8);
      acc0 = wmma_bf16(a, b0, acc0);
      acc1 = wmma_bf16(a, b1, acc1);
      acc2 = wmma_bf16(a, b2, acc2);
      acc3 = wmma_bf16(a, b3, acc3);
    }
  }

  float bv0 = bias[cobase + m];
  float bv1 = bias[cobase + 16 + m];
  float bv2 = bias[cobase + 32 + m];
  float bv3 = bias[cobase + 48 + m];
#pragma unroll
  for (int r = 0; r < 8; ++r) {
    int pr = (pt << 4) + r + half * 8;                // C/D layout: VGPR r -> M = r + 8*half
    uint16_t* orow = Ob + (size_t)pr * CCH + cobase + m;
    float v0 = acc0[r] + bv0, v1 = acc1[r] + bv1;
    float v2 = acc2[r] + bv2, v3 = acc3[r] + bv3;
    if (RELU) { v0 = fmaxf(v0, 0.f); v1 = fmaxf(v1, 0.f); v2 = fmaxf(v2, 0.f); v3 = fmaxf(v3, 0.f); }
    orow[0]  = f2bf(v0);
    orow[16] = f2bf(v1);
    orow[32] = f2bf(v2);
    orow[48] = f2bf(v3);
  }
}

// ---------------- cls_score conv + fused focal loss ----------------
// block = 16px tile (blockIdx.x) x 4 channel tiles (blockIdx.y*4 + wave), nCt = 45.

__global__ __launch_bounds__(128)
void k_cls_loss(const uint16_t* __restrict__ Ab, const uint16_t* __restrict__ Wt,
                const float* __restrict__ bias, const int* __restrict__ labels,
                float* __restrict__ out, int H, int W, int levelOff, int nTot) {
  __shared__ uint16_t smem[16 * CCH];
  const int HW = H * W;
  const int KTOT = 9 * CCH;
  const int nCt = (NUM_ANCHORS * NUM_CLASSES) / 16;   // 45
  int pt   = blockIdx.x;
  int tid  = threadIdx.x;
  int wave = tid >> 5, lane = tid & 31;
  int m = lane & 15, half = lane >> 4;
  int ct = blockIdx.y * 4 + wave;
  bool active = ct < nCt;                             // wave-uniform
  int cte = active ? ct : 0;

  int sb[4], sy[4], sx[4];
#pragma unroll
  for (int k = 0; k < 4; ++k) {
    int c = tid + k * 128;
    int p = (pt << 4) + (c >> 5);
    sb[k] = p / HW; int rem = p % HW; sy[k] = rem / W; sx[k] = rem % W;
  }

  v8f acc = {};
  const uint16_t* wbase = Wt + (size_t)(cte * 16 + m) * KTOT + half * 16;

  for (int t = 0; t < 9; ++t) {
    int dy = (t / 3) - 1, dx = (t % 3) - 1;
    __syncthreads();
#pragma unroll
    for (int k = 0; k < 4; ++k) {
      int c   = tid + k * 128;
      int off = (c & 31) * 8;
      int px  = c >> 5;
      int yy = sy[k] + dy, xx = sx[k] + dx;
      uint4 val = make_uint4(0u, 0u, 0u, 0u);
      if (yy >= 0 && yy < H && xx >= 0 && xx < W)
        val = *(const uint4*)(Ab + ((size_t)sb[k] * HW + (size_t)yy * W + xx) * CCH + off);
      *(uint4*)(smem + px * CCH + off) = val;
    }
    __syncthreads();
    if (active) {
      const uint16_t* wt = wbase + t * CCH;
#pragma unroll
      for (int kc = 0; kc < CCH; kc += 32) {
        Frag16 a;
        a.q[0] = *(const uint4*)(smem + m * CCH + kc + half * 16);
        a.q[1] = *(const uint4*)(smem + m * CCH + kc + half * 16 + 8);
        const uint16_t* wp = wt + kc;
        Frag16 b; b.q[0] = *(const uint4*)wp; b.q[1] = *(const uint4*)(wp + 8);
        acc = wmma_bf16(a, b, acc);
      }
    }
  }

  if (!active) return;
  int c = ct * 16 + m;
  float bv = bias[c];
  int aidx = c / NUM_CLASSES;
  int kcls = c % NUM_CLASSES;
  float lsum = 0.f;
#pragma unroll
  for (int r = 0; r < 8; ++r) {
    int pr = (pt << 4) + r + half * 8;
    int bi = pr / HW, hw = pr % HW;
    int n  = levelOff + hw * NUM_ANCHORS + aidx;
    int lab = labels[(size_t)bi * nTot + n];
    if (lab >= 0) {
      float l  = acc[r] + bv;
      float tt = (lab == kcls) ? 1.f : 0.f;
      float ce = fmaxf(l, 0.f) - l * tt + log1pf(__expf(-fabsf(l)));
      float pp = 1.f / (1.f + __expf(-l));
      float ptv = pp * tt + (1.f - pp) * (1.f - tt);
      float om  = 1.f - ptv;
      float at  = 0.25f * tt + 0.75f * (1.f - tt);
      lsum += at * ce * om * om;
    }
  }
  for (int o = 16; o > 0; o >>= 1) lsum += __shfl_down(lsum, o);
  if (lane == 0) atomicAdd(out + 0, lsum * 0.01f);
}

// ---------------- box_pred conv + fused L1 loss (CoutPad=48, 3 tiles) ----------------

__global__ __launch_bounds__(128)
void k_box_loss(const uint16_t* __restrict__ Ab, const uint16_t* __restrict__ Wt,
                const float* __restrict__ bias, const int* __restrict__ labels,
                const float* __restrict__ tdeltas, float* __restrict__ out,
                int H, int W, int levelOff, int nTot) {
  __shared__ uint16_t smem[16 * CCH];
  const int HW = H * W;
  const int KTOT = 9 * CCH;
  const int nCt = 3, CoutReal = 36;
  int pt   = blockIdx.x;
  int tid  = threadIdx.x;
  int wave = tid >> 5, lane = tid & 31;
  int m = lane & 15, half = lane >> 4;
  int ct = wave;
  bool active = ct < nCt;                             // wave 3 idles (barriers only)
  int cte = active ? ct : 0;

  int sb[4], sy[4], sx[4];
#pragma unroll
  for (int k = 0; k < 4; ++k) {
    int c = tid + k * 128;
    int p = (pt << 4) + (c >> 5);
    sb[k] = p / HW; int rem = p % HW; sy[k] = rem / W; sx[k] = rem % W;
  }

  v8f acc = {};
  const uint16_t* wbase = Wt + (size_t)(cte * 16 + m) * KTOT + half * 16;

  for (int t = 0; t < 9; ++t) {
    int dy = (t / 3) - 1, dx = (t % 3) - 1;
    __syncthreads();
#pragma unroll
    for (int k = 0; k < 4; ++k) {
      int c   = tid + k * 128;
      int off = (c & 31) * 8;
      int px  = c >> 5;
      int yy = sy[k] + dy, xx = sx[k] + dx;
      uint4 val = make_uint4(0u, 0u, 0u, 0u);
      if (yy >= 0 && yy < H && xx >= 0 && xx < W)
        val = *(const uint4*)(Ab + ((size_t)sb[k] * HW + (size_t)yy * W + xx) * CCH + off);
      *(uint4*)(smem + px * CCH + off) = val;
    }
    __syncthreads();
    if (active) {
      const uint16_t* wt = wbase + t * CCH;
#pragma unroll
      for (int kc = 0; kc < CCH; kc += 32) {
        Frag16 a;
        a.q[0] = *(const uint4*)(smem + m * CCH + kc + half * 16);
        a.q[1] = *(const uint4*)(smem + m * CCH + kc + half * 16 + 8);
        const uint16_t* wp = wt + kc;
        Frag16 b; b.q[0] = *(const uint4*)wp; b.q[1] = *(const uint4*)(wp + 8);
        acc = wmma_bf16(a, b, acc);
      }
    }
  }

  if (!active) return;
  int c = ct * 16 + m;
  float lsum = 0.f;
  if (c < CoutReal) {
    float bv = bias[c];
    int aidx = c / 4, comp = c % 4;
#pragma unroll
    for (int r = 0; r < 8; ++r) {
      int pr = (pt << 4) + r + half * 8;
      int bi = pr / HW, hw = pr % HW;
      int n  = levelOff + hw * NUM_ANCHORS + aidx;
      int lab = labels[(size_t)bi * nTot + n];
      if (lab >= 0 && lab != NUM_CLASSES) {
        float d = acc[r] + bv;
        lsum += fabsf(d - tdeltas[((size_t)bi * nTot + n) * 4 + comp]);
      }
    }
  }
  for (int o = 16; o > 0; o >>= 1) lsum += __shfl_down(lsum, o);
  if (lane == 0) atomicAdd(out + 1, lsum * 0.01f);
}

// ---------------- host orchestration ----------------

extern "C" void kernel_launch(void* const* d_in, const int* in_sizes, int n_in,
                              void* d_out, int out_size, void* d_ws, size_t ws_size,
                              hipStream_t stream) {
  (void)in_sizes; (void)n_in; (void)out_size; (void)ws_size;
  const float* feats[5] = {(const float*)d_in[0], (const float*)d_in[1],
                           (const float*)d_in[2], (const float*)d_in[3],
                           (const float*)d_in[4]};
  const int*   labels      = (const int*)d_in[5];
  const float* tdeltas     = (const float*)d_in[6];
  const float* in_proj_w   = (const float*)d_in[7];
  const float* in_proj_b   = (const float*)d_in[8];
  const float* cls_convs_w = (const float*)d_in[9];
  const float* cls_convs_b = (const float*)d_in[10];
  const float* cls_score_w = (const float*)d_in[11];
  const float* cls_score_b = (const float*)d_in[12];
  const float* box_convs_w = (const float*)d_in[13];
  const float* box_convs_b = (const float*)d_in[14];
  const float* box_pred_w  = (const float*)d_in[15];
  const float* box_pred_b  = (const float*)d_in[16];
  float* out = (float*)d_out;

  uint16_t* ws = (uint16_t*)d_ws;
  size_t off = 0;
  uint16_t* wInproj = ws + off; off += (size_t)5 * CCH * CCH;
  uint16_t* wCls    = ws + off; off += (size_t)4 * CCH * 9 * CCH;
  uint16_t* wBox    = ws + off; off += (size_t)4 * CCH * 9 * CCH;
  uint16_t* wCsc    = ws + off; off += (size_t)720 * 9 * CCH;
  uint16_t* wBpd    = ws + off; off += (size_t)48 * 9 * CCH;
  const size_t actElems = (size_t)2 * 128 * 128 * CCH;
  uint16_t* bufA = ws + off; off += actElems;
  uint16_t* bufX = ws + off; off += actElems;
  uint16_t* bufC = ws + off; off += actElems;

  hipLaunchKernelGGL(k_zero2, dim3(1), dim3(32), 0, stream, out);
  {
    int n = 5 * CCH * CCH;
    hipLaunchKernelGGL(k_cvt_inproj, dim3((n + 255) / 256), dim3(256), 0, stream, in_proj_w, wInproj);
    n = 4 * CCH * 9 * CCH;
    hipLaunchKernelGGL(k_cvt_convw, dim3((n + 255) / 256), dim3(256), 0, stream, cls_convs_w, wCls);
    hipLaunchKernelGGL(k_cvt_convw, dim3((n + 255) / 256), dim3(256), 0, stream, box_convs_w, wBox);
    n = 720 * 9 * CCH;
    hipLaunchKernelGGL(k_cvt_predw, dim3((n + 255) / 256), dim3(256), 0, stream, cls_score_w, wCsc, 720, 720);
    n = 48 * 9 * CCH;
    hipLaunchKernelGGL(k_cvt_predw, dim3((n + 255) / 256), dim3(256), 0, stream, box_pred_w, wBpd, 36, 48);
  }

  const int Hs[5] = {128, 64, 32, 16, 8};
  const int levelOff[5] = {0, 147456, 184320, 193536, 195840};
  const int nTot = 196416;

  for (int l = 0; l < 5; ++l) {
    int H = Hs[l], W = Hs[l], HW = H * W, pixTotal = 2 * HW;
    int nPt = pixTotal / 16;
    hipLaunchKernelGGL(k_f32_to_bf16, dim3(1024), dim3(256), 0, stream,
                       feats[l], bufA, pixTotal * CCH);
    hipLaunchKernelGGL((k_conv_wmma<false, 1>), dim3(nPt), dim3(128), 0, stream,
                       bufA, wInproj + (size_t)l * CCH * CCH, in_proj_b + l * CCH,
                       bufX, H, W);
    uint16_t* dsts[4] = {bufA, bufC, bufA, bufC};
    // cls branch
    const uint16_t* cur = bufX;
    for (int j = 0; j < 4; ++j) {
      hipLaunchKernelGGL((k_conv_wmma<true, 9>), dim3(nPt), dim3(128), 0, stream,
                         cur, wCls + (size_t)j * CCH * 9 * CCH, cls_convs_b + j * CCH,
                         dsts[j], H, W);
      cur = dsts[j];
    }
    hipLaunchKernelGGL(k_cls_loss, dim3(nPt, 12), dim3(128), 0, stream,
                       cur, wCsc, cls_score_b, labels, out, H, W, levelOff[l], nTot);
    // box branch (bufX is still intact)
    cur = bufX;
    for (int j = 0; j < 4; ++j) {
      hipLaunchKernelGGL((k_conv_wmma<true, 9>), dim3(nPt), dim3(128), 0, stream,
                         cur, wBox + (size_t)j * CCH * 9 * CCH, box_convs_b + j * CCH,
                         dsts[j], H, W);
      cur = dsts[j];
    }
    hipLaunchKernelGGL(k_box_loss, dim3(nPt, 1), dim3(128), 0, stream,
                       cur, wBpd, box_pred_b, labels, tdeltas, out, H, W, levelOff[l], nTot);
  }
}
